// LVCBlock_23991687316184
// MI455X (gfx1250) — matrix operand order repaired and checked by reading, not compiled, not run on hardware
//
#include <hip/hip_runtime.h>
#include <hip/hip_bf16.h>
#include <cstddef>

// ---------------------------------------------------------------------------
// UnivNet LVC block for MI455X (gfx1250, wave32, WMMA + async LDS copies).
// f16 multiplies, f32 WMMA accumulation; f16 inter-layer intermediates.
// ---------------------------------------------------------------------------

typedef _Float16 half16 __attribute__((ext_vector_type(16)));
typedef float    floatx8 __attribute__((ext_vector_type(8)));

#define N_LAYERS 10
#define KSZ      3
#define HOP      256
#define C_IN     64
#define C_OUT    128
#define C_COND   100
#define HID      64
#define LSEQ     16384
#define NB       4
#define NT       64                        // windows == cond timesteps
#define RPL      (C_IN * C_OUT * KSZ)      // rows per layer = 24576

// ---- A-fragment K order used here: contiguous k = 16*laneHi + hi ----------
__device__ __forceinline__ int kmapA(int hi, int laneHi) {
    return hi + (laneHi ? 16 : 0);
}

// ---- gfx1250 async global->LDS copy (ASYNCcnt-tracked, 16B per lane) ------
__device__ __forceinline__ void async_copy_b128(void* lds_dst, const void* gsrc) {
    unsigned l = (unsigned)(unsigned long long)lds_dst;   // low 32b = LDS offset
    asm volatile("global_load_async_to_lds_b128 %0, %1, off"
                 :: "v"(l), "v"(gsrc) : "memory");
}
__device__ __forceinline__ void wait_async0() {
    asm volatile("s_wait_asynccnt 0x0" ::: "memory");
}

// ===========================================================================
// Kernel 1: kernel-predictor hidden state h (4 x 64 x 64).
// H (f32, [hid][b*64+t]) for bias dots; Hh (f16, [(b,t)][hid]) for B-frags.
// ===========================================================================
__global__ void kp_hidden(const float* __restrict__ c,
                          const float* __restrict__ w_in, const float* __restrict__ b_in,
                          const float* __restrict__ w_r1, const float* __restrict__ b_r1,
                          const float* __restrict__ w_r2, const float* __restrict__ b_r2,
                          float* __restrict__ H, _Float16* __restrict__ Hh)
{
    __shared__ float cb[C_COND][NT + 4];
    __shared__ float h0[NT][HID];          // [t][ch]: conflict-free
    __shared__ float r1[NT][HID];

    const int b = blockIdx.x;
    const int tid = threadIdx.x;

    for (int idx = tid; idx < C_COND * (NT + 4); idx += 256)
        cb[idx / (NT + 4)][idx % (NT + 4)] = 0.0f;
    __syncthreads();
    for (int idx = tid; idx < C_COND * NT; idx += 256) {
        int ic = idx >> 6, t = idx & 63;
        cb[ic][t + 2] = c[((size_t)b * C_COND + ic) * NT + t];
    }
    __syncthreads();

    const int ch = tid & 63, tg = tid >> 6;

    for (int jj = 0; jj < 16; ++jj) {           // conv_in k=5 pad=2 + lrelu
        int t = tg * 16 + jj;
        float acc = b_in[ch];
        for (int ic = 0; ic < C_COND; ++ic) {
            const float* wrow = w_in + (ch * C_COND + ic) * 5;
#pragma unroll
            for (int k = 0; k < 5; ++k) acc += cb[ic][t + k] * wrow[k];
        }
        h0[t][ch] = (acc >= 0.0f) ? acc : 0.1f * acc;
    }
    __syncthreads();

    for (int jj = 0; jj < 16; ++jj) {           // r1 = lrelu(Wr1 h0 + br1)
        int t = tg * 16 + jj;
        float acc = b_r1[ch];
        for (int ic = 0; ic < HID; ++ic) acc += h0[t][ic] * w_r1[ch * HID + ic];
        r1[t][ch] = (acc >= 0.0f) ? acc : 0.1f * acc;
    }
    __syncthreads();

    for (int jj = 0; jj < 16; ++jj) {           // h = h0 + Wr2 r1 + br2
        int t = tg * 16 + jj;
        float acc = b_r2[ch];
        for (int ic = 0; ic < HID; ++ic) acc += r1[t][ic] * w_r2[ch * HID + ic];
        float v = h0[t][ch] + acc;
        H[ch * (NB * NT) + b * NT + t] = v;
        Hh[(size_t)(b * NT + t) * HID + ch] = (_Float16)v;
    }
}

// ===========================================================================
// Kernel 2 (per layer): predicted-weights GEMM via WMMA.
//   k[24576 x 256(b,t)] = w_k(layer)[24576 x 64] * H[64 x 256] + b_k
// Output is f16, per-column, ALREADY in WMMA A-fragment lane order so the LVC
// kernel can async-copy it byte-for-byte into LDS.
// ===========================================================================
__global__ void predict_weights(const float* __restrict__ w_k, const float* __restrict__ b_k,
                                const _Float16* __restrict__ Hh, _Float16* __restrict__ Wlh,
                                int layer)
{
    const int tid  = threadIdx.x;
    const int wave = tid >> 5, lane = tid & 31;
    const int ln = lane & 15, lhi = (lane >> 4) & 1;

    const int rBase = blockIdx.x * 128 + wave * 16;
    const size_t rowG = (size_t)(layer * RPL + rBase + ln) * HID;

    // A fragments (K = 64 in 2 steps of 32), f32 -> f16
    half16 a0, a1;
#pragma unroll
    for (int hi = 0; hi < 16; ++hi) {
        a0[hi] = (_Float16)w_k[rowG + kmapA(hi, lhi)];
        a1[hi] = (_Float16)w_k[rowG + 32 + kmapA(hi, lhi)];
    }

    // ct-independent: output fragment position + bias for this lane's 8 rows
    int   fragpos[8];
    float bkv[8];
#pragma unroll
    for (int j = 0; j < 8; ++j) {
        int r  = rBase + j + (lhi ? 8 : 0);
        int kk = r % 3, t2 = r / 3;
        int co = t2 & 127, ci = t2 >> 7;
        int k  = kk * 64 + ci;
        int ks = k >> 5, lk = k & 31;
        int grp, hi;
        grp = lk >> 4; hi = lk & 15;
        fragpos[j] = (((ks << 3) + (co >> 4)) * 32 + ((co & 15) | (grp << 4))) * 16 + hi;
        bkv[j] = b_k[layer * RPL + r];
    }

#pragma unroll 4
    for (int ct = 0; ct < 16; ++ct) {
        const _Float16* hb = Hh + (size_t)(ct * 16 + ln) * HID + (lhi ? 16 : 0);
        half16 b0 = *(const half16*)hb;
        half16 b1 = *(const half16*)(hb + 32);
        floatx8 acc = {};
        acc = __builtin_amdgcn_wmma_f32_16x16x32_f16(false, a0, false, b0,
                                                     (short)0, acc, false, false);
        acc = __builtin_amdgcn_wmma_f32_16x16x32_f16(false, a1, false, b1,
                                                     (short)0, acc, false, false);
        int col = ct * 16 + ln;
        _Float16* out = Wlh + (size_t)col * RPL;
#pragma unroll
        for (int j = 0; j < 8; ++j)
            out[fragpos[j]] = (_Float16)(acc[j] + bkv[j]);
    }
}

// ===========================================================================
// Kernel 3 (per layer): fused LVC + gated activation.
// grid (window d, batch b); block 256 = 8 waves; each wave owns 2 s-tiles x
// all 8 co-tiles so gating pairs (co, co+64) stay in registers.
// Weight fragments arrive via GLOBAL_LOAD_ASYNC_TO_LDS_B128 (48 KB, byte
// copy — already fragment-ordered); x window staged with f32/f16->f16
// transpose into [tap][s][ci].  LDS total: 96 + 48 + 0.5 KB = 145 KB.
// ===========================================================================
template <typename TIN, typename TOUT>
__global__ void lvc_layer(const TIN* __restrict__ xin, const _Float16* __restrict__ Wlh,
                          const float* __restrict__ H,  const float* __restrict__ w_b,
                          const float* __restrict__ b_b, TOUT* __restrict__ xout,
                          int layer, int dil)
{
    __shared__ __align__(32) _Float16 xT[3 * HOP * C_IN];        // [tap][s][ci]
    __shared__ __align__(32) _Float16 Wfrag[6 * 8 * 32 * 16];    // A-frag order
    __shared__ float biasS[C_OUT];

    const int d = blockIdx.x, b = blockIdx.y;
    const int bt = b * NT + d;
    const int tid = threadIdx.x;

    // ---- async-stage weight fragments (contiguous 48 KB) ------------------
    {
        const _Float16* gW = Wlh + (size_t)bt * RPL;
#pragma unroll
        for (int it = 0; it < RPL / (256 * 8); ++it) {           // 12 x 16B/thread
            int off = (it * 256 + tid) * 8;
            async_copy_b128(&Wfrag[off], gW + off);
        }
    }

    // ---- stage x window: 3 taps x 256 s x 64 ci, transpose + cvt to f16 ---
    for (int idx = tid; idx < 3 * C_IN * HOP; idx += 256) {
        int s  = idx & 255;
        int ci = (idx >> 8) & 63;
        int kk = idx >> 14;
        int pos = d * HOP + s + (kk - 1) * dil;
        float v = (pos >= 0 && pos < LSEQ)
                      ? (float)xin[((size_t)b * C_IN + ci) * LSEQ + pos] : 0.0f;
        xT[(kk * HOP + s) * C_IN + ci] = (_Float16)v;
    }

    // ---- per-window bias from hidden state --------------------------------
    if (tid < C_OUT) {
        int ch = layer * C_OUT + tid;
        float acc = b_b[ch];
        for (int hc = 0; hc < HID; ++hc)
            acc += w_b[ch * HID + hc] * H[hc * (NB * NT) + bt];
        biasS[tid] = acc;
    }

    wait_async0();           // drain this wave's async copies
    __syncthreads();         // all waves' LDS contributions visible

    // ---- WMMA main loop: O[128 x 256] = W[128 x 192] * X[192 x 256] -------
    const int wave = tid >> 5, lane = tid & 31;
    const int ln = lane & 15, lhi = (lane >> 4) & 1;

    for (int stl = 0; stl < 2; ++stl) {
        const int st = wave * 2 + stl;
        floatx8 zero = {};
        floatx8 acc[8];
#pragma unroll
        for (int i = 0; i < 8; ++i) acc[i] = zero;

#pragma unroll
        for (int ks = 0; ks < 6; ++ks) {
            int Kg0 = ks * 32 + (lhi ? 16 : 0);
            int kk  = Kg0 >> 6;
            int ci  = Kg0 & 63;
            half16 bfrag = *(const half16*)&xT[((kk * HOP) + st * 16 + ln) * C_IN + ci];
#pragma unroll
            for (int cot = 0; cot < 8; ++cot) {
                half16 afrag = *(const half16*)&Wfrag[(((ks << 3) + cot) * 32 + lane) * 16];
                acc[cot] = __builtin_amdgcn_wmma_f32_16x16x32_f16(
                    false, afrag, false, bfrag, (short)0, acc[cot], false, false);
            }
        }

        // ---- bias + gated activation, write x_out -------------------------
        const int sOut = d * HOP + st * 16 + ln;
        TOUT* outBase = xout + (size_t)b * C_IN * LSEQ + sOut;
#pragma unroll
        for (int cot = 0; cot < 4; ++cot) {
#pragma unroll
            for (int j = 0; j < 8; ++j) {
                int coG = cot * 16 + j + (lhi ? 8 : 0);
                float g  = acc[cot][j]     + biasS[coG];
                float tv = acc[cot + 4][j] + biasS[coG + 64];
                float sg = 1.0f / (1.0f + __expf(-g));
                outBase[(size_t)coG * LSEQ] = (TOUT)(sg * tanhf(tv));
            }
        }
    }
}

// ===========================================================================
// Host launcher
// ===========================================================================
extern "C" void kernel_launch(void* const* d_in, const int* in_sizes, int n_in,
                              void* d_out, int out_size, void* d_ws, size_t ws_size,
                              hipStream_t stream)
{
    const float* x    = (const float*)d_in[0];
    const float* c    = (const float*)d_in[1];
    const float* w_in = (const float*)d_in[2];
    const float* b_in = (const float*)d_in[3];
    const float* w_r1 = (const float*)d_in[4];
    const float* b_r1 = (const float*)d_in[5];
    const float* w_r2 = (const float*)d_in[6];
    const float* b_r2 = (const float*)d_in[7];
    const float* w_k  = (const float*)d_in[8];
    const float* b_k  = (const float*)d_in[9];
    const float* w_b  = (const float*)d_in[10];
    const float* b_b  = (const float*)d_in[11];

    // workspace carve-up (~29 MB total)
    char* ws = (char*)d_ws;
    float*    H   = (float*)ws;                                  // 64 KB
    _Float16* Hh  = (_Float16*)(ws + 65536);                     // 32 KB
    _Float16* Wlh = (_Float16*)(ws + 65536 + 32768);             // 24576*256*2 = 12 MB
    _Float16* xA  = (_Float16*)(ws + 65536 + 32768 + (size_t)RPL * NB * NT * 2);
    _Float16* xB  = xA + (size_t)NB * C_IN * LSEQ;               // 8 MB each

    kp_hidden<<<dim3(NB), dim3(256), 0, stream>>>(c, w_in, b_in, w_r1, b_r1,
                                                  w_r2, b_r2, H, Hh);

    const dim3 g(NT, NB), blk(256);
    for (int layer = 0; layer < N_LAYERS; ++layer) {
        predict_weights<<<dim3(RPL / 128), blk, 0, stream>>>(w_k, b_k, Hh, Wlh, layer);

        if (layer == 0) {
            lvc_layer<float, _Float16><<<g, blk, 0, stream>>>(
                x, Wlh, H, w_b, b_b, xA, 0, 1);
        } else if (layer == N_LAYERS - 1) {
            const _Float16* in = (layer & 1) ? xA : xB;
            lvc_layer<_Float16, float><<<g, blk, 0, stream>>>(
                in, Wlh, H, w_b, b_b, (float*)d_out, layer, 1 << layer);
        } else {
            const _Float16* in  = (layer & 1) ? xA : xB;
            _Float16*       out = (layer & 1) ? xB : xA;
            lvc_layer<_Float16, _Float16><<<g, blk, 0, stream>>>(
                in, Wlh, H, w_b, b_b, out, layer, 1 << layer);
        }
    }
}
